// AttentionHead_30734785970804
// MI455X (gfx1250) — matrix operand order repaired and checked by reading, not compiled, not run on hardware
//
#include <hip/hip_runtime.h>
#include <hip/hip_bf16.h>

// ---------------------------------------------------------------------------
// AttentionHead: out = softmax(mask(qWq @ (kWk)^T)/sqrt(64)) @ (vWv)
// B=4, T=4096, D=1024, H=64.  fp32 in/out, f16 WMMA internally (f32 accum).
// ---------------------------------------------------------------------------

typedef __attribute__((ext_vector_type(16))) _Float16 v16h;
typedef __attribute__((ext_vector_type(8)))  _Float16 v8h;
typedef __attribute__((ext_vector_type(8)))  float    v8f;
typedef __attribute__((ext_vector_type(4)))  float    v4f;

#define BATCH 4
#define TSEQ  4096
#define DMODEL 1024
#define HEAD  64
#define BT    (BATCH * TSEQ)

__device__ __forceinline__ v8f wmma_f16(v16h a, v16h b, v8f c) {
  // D = A(16x32 f16) * B(32x16 f16) + C(16x16 f32)
  return __builtin_amdgcn_wmma_f32_16x16x32_f16(
      /*neg_a=*/false, a, /*neg_b=*/false, b,
      /*c_mod=*/(short)0, c, /*reuse_a=*/false, /*reuse_b=*/false);
}

// ---------------------------------------------------------------------------
// Kernel 1: convert Wq/Wk/Wv (fp32 [1024][64]) into f16 B-fragment layout.
// Fragment f = (ktile*4 + ntile), 32 lanes x 16 halves each, stored contiguous:
//   wz[w][((f*32 + lane)*16) + h] = W[ ktile*32 + (h<8 ? 8*(lane/16)+h : 8+8*(lane/16)+h) ][ ntile*16 + lane%16 ]
// ---------------------------------------------------------------------------
__global__ void wswz_kernel(const float* __restrict__ Wq,
                            const float* __restrict__ Wk,
                            const float* __restrict__ Wv,
                            _Float16* __restrict__ wz) {
  int idx = blockIdx.x * 256 + threadIdx.x;     // 3 * 65536 total
  int w   = idx >> 16;
  int rem = idx & 65535;
  int f   = rem >> 9;                            // 0..127  (ktile*4 + ntile)
  int l   = (rem >> 4) & 31;
  int h   = rem & 15;
  int t   = f >> 2, n = f & 3;
  int K   = t * 32 + ((h < 8) ? (8 * (l >> 4) + h) : (16 + 8 * (l >> 4) + h - 8));
  int N   = n * 16 + (l & 15);
  const float* W = (w == 0) ? Wq : (w == 1) ? Wk : Wv;
  wz[(size_t)w * 65536 + rem] = (_Float16)W[K * HEAD + N];
}

// ---------------------------------------------------------------------------
// Kernel 2: projections.  One wave computes a 16x64 output tile.
//   which=0: qh = q@Wq  (row-major f16 [BT][64])
//   which=1: kh = k@Wk  (row-major f16 [BT][64])
//   which=2: vhT = (v@Wv)^T per batch (f16 [B][64][T])
// ---------------------------------------------------------------------------
__global__ void proj_kernel(const float* __restrict__ q,
                            const float* __restrict__ k,
                            const float* __restrict__ v,
                            const _Float16* __restrict__ wz,
                            _Float16* __restrict__ qh,
                            _Float16* __restrict__ kh,
                            _Float16* __restrict__ vhT) {
  int which = blockIdx.y;
  const float*    x  = (which == 0) ? q : (which == 1) ? k : v;
  const _Float16* wzp = wz + (size_t)which * 65536;

  int wv   = threadIdx.x >> 5;
  int lane = threadIdx.x & 31;
  int g    = lane >> 4;          // lane half
  int rho  = lane & 15;          // row within tile / column id
  int tile = blockIdx.x * 8 + wv;  // 0..1023
  int row0 = tile * 16;

  const float* xrow = x + (size_t)(row0 + rho) * DMODEL;

  v8f acc[4];
  for (int n = 0; n < 4; ++n) acc[n] = v8f{0.f,0.f,0.f,0.f,0.f,0.f,0.f,0.f};

  for (int kk = 0; kk < 32; ++kk) {             // K = 1024, 32 per step
    int k0 = kk * 32;
    // Speculative prefetch ~4 iterations ahead of the HBM stream
    // (global_prefetch_b8: no counter traffic, whole cacheline pulled).
    if (kk < 28) __builtin_prefetch(xrow + k0 + 128, 0, 1);
    // A fragment: lane holds row rho, cols {k0+8g..+7} and {k0+16+8g..+7}
    v4f x0 = *(const v4f*)(xrow + k0 + 8 * g);
    v4f x1 = *(const v4f*)(xrow + k0 + 8 * g + 4);
    v4f x2 = *(const v4f*)(xrow + k0 + 16 + 8 * g);
    v4f x3 = *(const v4f*)(xrow + k0 + 16 + 8 * g + 4);
    v16h a;
    #pragma unroll
    for (int i = 0; i < 4; ++i) {
      a[i]      = (_Float16)x0[i];
      a[4 + i]  = (_Float16)x1[i];
      a[8 + i]  = (_Float16)x2[i];
      a[12 + i] = (_Float16)x3[i];
    }
    #pragma unroll
    for (int n = 0; n < 4; ++n) {
      const v8h* bp = (const v8h*)(wzp + ((size_t)(kk * 4 + n) * 32 + lane) * 16);
      v8h blo = bp[0], bhi = bp[1];
      v16h b;
      #pragma unroll
      for (int i = 0; i < 8; ++i) { b[i] = blo[i]; b[8 + i] = bhi[i]; }
      acc[n] = wmma_f16(a, b, acc[n]);
    }
  }

  // Store (C layout: reg r -> row r + 8g, col = rho within n-tile)
  if (which < 2) {
    _Float16* out = (which == 0) ? qh : kh;
    #pragma unroll
    for (int n = 0; n < 4; ++n)
      #pragma unroll
      for (int r = 0; r < 8; ++r)
        out[(size_t)(row0 + r + 8 * g) * HEAD + n * 16 + rho] = (_Float16)acc[n][r];
  } else {
    int b  = row0 >> 12;            // /TSEQ
    int tl = row0 & (TSEQ - 1);
    #pragma unroll
    for (int n = 0; n < 4; ++n)
      #pragma unroll
      for (int r = 0; r < 8; ++r)
        vhT[((size_t)b * HEAD + n * 16 + rho) * TSEQ + tl + r + 8 * g] =
            (_Float16)acc[n][r];
  }
}

// ---------------------------------------------------------------------------
// Kernel 3: flash attention.  One wave owns 16 query rows; kv in blocks of 32.
// ---------------------------------------------------------------------------
__global__ void attn_kernel(const _Float16* __restrict__ qh,
                            const _Float16* __restrict__ kh,
                            const _Float16* __restrict__ vhT,
                            const unsigned char* __restrict__ mask,
                            float* __restrict__ out) {
  __shared__ float lds[8][16 * 36];             // per-wave 16x32 tile, padded

  int wv   = threadIdx.x >> 5;
  int lane = threadIdx.x & 31;
  int g    = lane >> 4;
  int rho  = lane & 15;
  int b    = blockIdx.y;
  int qt   = blockIdx.x * 8 + wv;               // 0..255 query tiles
  int q0   = qt * 16;                           // within batch
  int qrow = b * TSEQ + q0;                     // global row base

  float* myl = lds[wv];

  // Q A-fragments (d chunks 0..31, 32..63), loaded once.
  v16h aq[2];
  {
    const _Float16* qb = qh + (size_t)(qrow + rho) * HEAD;
    #pragma unroll
    for (int dc = 0; dc < 2; ++dc) {
      v8h lo0 = *(const v8h*)(qb + dc * 32 + 8 * g);
      v8h hi0 = *(const v8h*)(qb + dc * 32 + 16 + 8 * g);
      #pragma unroll
      for (int i = 0; i < 8; ++i) { aq[dc][i] = lo0[i]; aq[dc][8 + i] = hi0[i]; }
    }
  }

  v8f oc[4];
  #pragma unroll
  for (int n = 0; n < 4; ++n) oc[n] = v8f{0.f,0.f,0.f,0.f,0.f,0.f,0.f,0.f};
  float m_row = -1e30f;   // stats for row rho (replicated in both halves)
  float l_row = 0.0f;

  int nblk = (q0 + 47) >> 5;                    // causal: kv <= q0+15
  for (int kvb = 0; kvb < nblk; ++kvb) {
    int kv0 = kvb * 32;

    // Warm next block's K rows / V^T runs while this block does softmax.
    if (kvb + 1 < nblk) {
      __builtin_prefetch(
          kh + (size_t)(b * TSEQ + kv0 + 32 + (lane & 31)) * HEAD, 0, 1);
      __builtin_prefetch(
          vhT + ((size_t)b * HEAD + 2 * (lane & 31)) * TSEQ + kv0 + 32, 0, 1);
    }

    // ---- S = Q K^T for two 16-col tiles -------------------------------
    v8f s[2];
    #pragma unroll
    for (int ct = 0; ct < 2; ++ct) {
      s[ct] = v8f{0.f,0.f,0.f,0.f,0.f,0.f,0.f,0.f};
      const _Float16* kb =
          kh + (size_t)(b * TSEQ + kv0 + ct * 16 + rho) * HEAD;
      #pragma unroll
      for (int dc = 0; dc < 2; ++dc) {
        // B frag: lane rho = kv column, halves = d {8g..+7} and {16+8g..+7}
        v8h lo0 = *(const v8h*)(kb + dc * 32 + 8 * g);
        v8h hi0 = *(const v8h*)(kb + dc * 32 + 16 + 8 * g);
        v16h bk;
        #pragma unroll
        for (int i = 0; i < 8; ++i) { bk[i] = lo0[i]; bk[8 + i] = hi0[i]; }
        s[ct] = wmma_f16(aq[dc], bk, s[ct]);
      }
    }

    // ---- mask + scale, C-layout -> LDS [row][col] ---------------------
    #pragma unroll
    for (int ct = 0; ct < 2; ++ct) {
      #pragma unroll
      for (int r = 0; r < 8; ++r) {
        int qr = q0 + r + 8 * g;
        int kv = kv0 + ct * 16 + rho;
        float sv = s[ct][r] * 0.125f;           // 1/sqrt(64)
        unsigned char keep = mask[(size_t)qr * TSEQ + kv];
        sv = keep ? sv : -1e30f;
        myl[(r + 8 * g) * 36 + ct * 16 + rho] = sv;
      }
    }
    asm volatile("s_wait_dscnt 0" ::: "memory");

    // ---- A-orientation read: lane = row rho, its 16 of 32 cols --------
    float* rp = myl + rho * 36;
    v4f t0 = *(const v4f*)(rp + 8 * g);
    v4f t1 = *(const v4f*)(rp + 8 * g + 4);
    v4f t2 = *(const v4f*)(rp + 16 + 8 * g);
    v4f t3 = *(const v4f*)(rp + 16 + 8 * g + 4);
    float tv[16];
    #pragma unroll
    for (int i = 0; i < 4; ++i) {
      tv[i] = t0[i]; tv[4 + i] = t1[i]; tv[8 + i] = t2[i]; tv[12 + i] = t3[i];
    }

    // ---- online softmax ----------------------------------------------
    float mx = tv[0];
    #pragma unroll
    for (int i = 1; i < 16; ++i) mx = fmaxf(mx, tv[i]);
    mx = fmaxf(mx, __shfl_xor(mx, 16, 32));     // combine halves -> full row
    float m_new = fmaxf(m_row, mx);
    float fac = __expf(m_row - m_new);

    v16h ap;
    float sum = 0.f;
    #pragma unroll
    for (int i = 0; i < 16; ++i) {
      float p = __expf(tv[i] - m_new);          // masked lanes underflow to 0
      sum += p;
      ap[i] = (_Float16)p;
    }
    sum += __shfl_xor(sum, 16, 32);
    l_row = l_row * fac + sum;
    m_row = m_new;

    // ---- rescale O (row of reg r is r+8g; its stats live in lane r+8g)
    #pragma unroll
    for (int r = 0; r < 8; ++r) {
      float fr = __shfl(fac, r + 8 * g, 32);
      #pragma unroll
      for (int n = 0; n < 4; ++n) oc[n][r] *= fr;
    }

    // ---- O += P @ V ---------------------------------------------------
    #pragma unroll
    for (int n = 0; n < 4; ++n) {
      const _Float16* vb =
          vhT + ((size_t)b * HEAD + n * 16 + rho) * TSEQ + kv0 + 8 * g;
      v8h lo0 = *(const v8h*)(vb);
      v8h hi0 = *(const v8h*)(vb + 16);
      v16h bv;
      #pragma unroll
      for (int i = 0; i < 8; ++i) { bv[i] = lo0[i]; bv[8 + i] = hi0[i]; }
      oc[n] = wmma_f16(ap, bv, oc[n]);
    }
  }

  // ---- normalize + store fp32 out [B][T][64] --------------------------
  #pragma unroll
  for (int r = 0; r < 8; ++r) {
    float lr = __shfl(l_row, r + 8 * g, 32);
    float inv = 1.0f / lr;
    #pragma unroll
    for (int n = 0; n < 4; ++n)
      out[(size_t)(qrow + r + 8 * g) * HEAD + n * 16 + rho] = oc[n][r] * inv;
  }
}

// ---------------------------------------------------------------------------
// Host launcher
// ---------------------------------------------------------------------------
extern "C" void kernel_launch(void* const* d_in, const int* in_sizes, int n_in,
                              void* d_out, int out_size, void* d_ws, size_t ws_size,
                              hipStream_t stream) {
  const float* q  = (const float*)d_in[0];
  const float* k  = (const float*)d_in[1];
  const float* v  = (const float*)d_in[2];
  const float* Wq = (const float*)d_in[3];
  const float* Wk = (const float*)d_in[4];
  const float* Wv = (const float*)d_in[5];
  const unsigned char* mask = (const unsigned char*)d_in[6];
  float* out = (float*)d_out;

  char* ws = (char*)d_ws;
  _Float16* wz  = (_Float16*)ws;                          // 3*65536*2 = 384 KB
  _Float16* qh  = (_Float16*)(ws + 393216);               // 2 MB
  _Float16* kh  = (_Float16*)(ws + 393216 + 2097152);     // 2 MB
  _Float16* vhT = (_Float16*)(ws + 393216 + 2 * 2097152); // 2 MB

  wswz_kernel<<<768, 256, 0, stream>>>(Wq, Wk, Wv, wz);
  proj_kernel<<<dim3(128, 3), 256, 0, stream>>>(q, k, v, wz, qh, kh, vhT);
  attn_kernel<<<dim3(32, 4), 256, 0, stream>>>(qh, kh, vhT, mask, out);
}
